// ScaledDotProductAttention_17025250362026
// MI455X (gfx1250) — matrix operand order repaired and checked by reading, not compiled, not run on hardware
//
#include <hip/hip_runtime.h>
#include <math.h>

// CDNA5 / gfx1250 flash-attention (block-causal, packed docs).
// wave32; WMMA f32_16x16x32_f16 for QK^T and PV; fp32 online softmax.
// Fast path: one-time f32->f16 conversion prepass into workspace (K row-major,
// V transposed), then TDM (tensor_load_to_lds) double-buffered tile DMA into LDS.

typedef __attribute__((ext_vector_type(16))) _Float16     v16h;
typedef __attribute__((ext_vector_type(8)))  _Float16     v8h;
typedef __attribute__((ext_vector_type(8)))  float        v8f;
typedef __attribute__((ext_vector_type(4)))  unsigned int v4u;
typedef __attribute__((ext_vector_type(8)))  int          v8i;
typedef __attribute__((ext_vector_type(4)))  int          v4i;

#define S_LEN   2048
#define D_DIM   128
#define H_NUM   16
#define BM      64          // query rows per workgroup
#define BN      32          // kv positions per inner tile (= WMMA K for PV)
#define NDOC    4
#define NTHREADS 128        // 4 waves of 32

union V16 { v16h v; v8h h[2]; };

// ---------------------------------------------------------------------------
// TDM: 2D f16 tile DMA global->LDS.  D# per CDNA5 ISA ch.8:
//   group0: [1:0]=count=1, [63:32]=lds_addr, [120:64]=global_addr, [127:126]=type 2
//   group1: [17:16]=data_size(1=2B), dim0@[79:48], dim1@[111:80], tile0@[127:112],
//           tile1@[143:128], stride0@[207:160]
// 6-arg builtin form (clang-23 lane): (v4u, v8i, v4i, v4i, v8i, i32 cpol).
// ---------------------------------------------------------------------------
__device__ inline void tdm_load_2d_f16(unsigned int lds_off, const _Float16* gaddr,
                                       unsigned int tensor_d0, unsigned int tensor_d1,
                                       unsigned long long stride0_elems,
                                       unsigned int tile_d0, unsigned int tile_d1)
{
    const unsigned long long ga = (unsigned long long)(uintptr_t)gaddr;
    v4u g0;
    g0.x = 1u;                                                    // count=1
    g0.y = lds_off;                                               // lds_addr
    g0.z = (unsigned int)ga;                                      // global_addr[31:0]
    g0.w = (unsigned int)((ga >> 32) & 0x01FFFFFFull)             // global_addr[56:32]
         | (2u << 30);                                            // type=2 ("image")
    v8i g1;
    g1[0] = (int)(1u << 16);                                      // data_size = 2 bytes
    g1[1] = (int)((tensor_d0 & 0xFFFFu) << 16);                   // tensor_dim0[15:0]
    g1[2] = (int)(((tensor_d0 >> 16) & 0xFFFFu) | ((tensor_d1 & 0xFFFFu) << 16));
    g1[3] = (int)(((tensor_d1 >> 16) & 0xFFFFu) | ((tile_d0 & 0xFFFFu) << 16));
    g1[4] = (int)(tile_d1 & 0xFFFFu);                             // tile_dim1; tile_dim2=0
    g1[5] = (int)(unsigned int)(stride0_elems & 0xFFFFFFFFull);   // stride0[31:0]
    g1[6] = (int)((stride0_elems >> 32) & 0xFFFFull);             // stride0[47:32]
    g1[7] = 0;
    v4i z4 = {0, 0, 0, 0};
    v8i z8 = {0, 0, 0, 0, 0, 0, 0, 0};
    __builtin_amdgcn_tensor_load_to_lds(g0, g1, z4, z4, z8, 0);
}

// ---------------------------------------------------------------------------
// Prepass: K -> f16 row-major [h][s][d]; V -> f16 transposed [h][d][s].
// ---------------------------------------------------------------------------
__global__ __launch_bounds__(256)
void convert_kv_kernel(const float* __restrict__ k, const float* __restrict__ v,
                       _Float16* __restrict__ kws, _Float16* __restrict__ vtws)
{
    const size_t i = (size_t)blockIdx.x * 256 + threadIdx.x;   // over H*S*D
    const int    d  = (int)(i & (D_DIM - 1));
    const size_t hs = i >> 7;
    const int    s  = (int)(hs & (S_LEN - 1));
    const int    h  = (int)(hs >> 11);
    kws[i] = (_Float16)k[i];
    vtws[((size_t)h * D_DIM + d) * S_LEN + s] = (_Float16)v[i];
}

// ---------------------------------------------------------------------------
// Main kernel (TDM fast path): K/V tiles arrive via tensor DMA, double-buffered.
// ---------------------------------------------------------------------------
__global__ __launch_bounds__(NTHREADS)
void fa_tdm_kernel(const float* __restrict__ qg,
                   const _Float16* __restrict__ kws,
                   const _Float16* __restrict__ vtws,
                   const long long* __restrict__ offsets,
                   float* __restrict__ outg)
{
    __shared__ _Float16 qLds[BM][D_DIM];             // 16 KB
    __shared__ _Float16 kLds[2][BN][D_DIM];          // 16 KB double-buffered
    __shared__ _Float16 vtLds[2][D_DIM][BN];         // 16 KB double-buffered
    __shared__ _Float16 pLds[NTHREADS/32][16][BN];   //  4 KB per-wave P scratch

    const int tid  = threadIdx.x;
    const int wave = tid >> 5;
    const int lane = tid & 31;
    const int l16  = lane & 15;
    const int hi   = lane >> 4;

    const int head = blockIdx.y;
    const int m0   = blockIdx.x * BM;

    const float*    qh  = qg   + (size_t)head * S_LEN * D_DIM;
    const _Float16* khw = kws  + (size_t)head * S_LEN * D_DIM;
    const _Float16* vhw = vtws + (size_t)head * D_DIM * S_LEN;  // [d][s]
    float*          oh  = outg + (size_t)head * S_LEN * D_DIM;

    int off[NDOC + 1];
    #pragma unroll
    for (int i = 0; i <= NDOC; ++i) off[i] = (int)offsets[i];

    // stage Q (f32 -> f16) into LDS
    for (int e = tid; e < BM * D_DIM; e += NTHREADS) {
        const int r = e >> 7, c = e & (D_DIM - 1);
        qLds[r][c] = (_Float16)qh[(size_t)(m0 + r) * D_DIM + c];
    }

    // kv range
    int seg0 = 0;
    #pragma unroll
    for (int i = 1; i <= NDOC; ++i) if (m0 >= off[i]) seg0 = i;
    const int kv_lo  = off[seg0] & ~(BN - 1);
    const int kv_end = m0 + BM;
    const int ntiles = (kv_end - kv_lo) >> 5;

    // kick off tile 0 DMA (wave 0 issues; TENSORcnt is per-wave)
    if (wave == 0) {
        tdm_load_2d_f16((unsigned int)(uintptr_t)&kLds[0][0][0],
                        khw + (size_t)kv_lo * D_DIM,
                        D_DIM, S_LEN, D_DIM, /*tile*/ D_DIM, BN);
        tdm_load_2d_f16((unsigned int)(uintptr_t)&vtLds[0][0][0],
                        vhw + kv_lo,
                        S_LEN, D_DIM, S_LEN, /*tile*/ BN, D_DIM);
    }
    __syncthreads();   // qLds ready

    // Q A-fragments held in VGPRs for the whole kernel
    const int wrow0 = wave * 16;
    V16 qa[4];
    {
        const int row = wrow0 + l16;
        const int dof = hi ? 8 : 0;
        #pragma unroll
        for (int c = 0; c < 4; ++c) {
            qa[c].h[0] = *(const v8h*)&qLds[row][32 * c + dof];
            qa[c].h[1] = *(const v8h*)&qLds[row][32 * c + 16 + dof];
        }
    }

    const int rbase = m0 + wrow0 + 8 * hi;
    int segrow[8];
    #pragma unroll
    for (int r = 0; r < 8; ++r) {
        int p = rbase + r, s = 0;
        #pragma unroll
        for (int i = 1; i <= NDOC; ++i) if (p >= off[i]) s = i;
        segrow[r] = s;
    }

    const float NEG_INF = -__builtin_inff();
    float mrow[8], lrow[8];
    v8f oacc[8];
    #pragma unroll
    for (int r = 0; r < 8; ++r) { mrow[r] = NEG_INF; lrow[r] = 0.f; }
    #pragma unroll
    for (int dc = 0; dc < 8; ++dc)
        #pragma unroll
        for (int r = 0; r < 8; ++r) oacc[dc][r] = 0.f;

    const float scale = 0.08838834764831845f; // 1/sqrt(128)

    for (int t = 0; t < ntiles; ++t) {
        const int kv  = kv_lo + t * BN;
        const int cur = t & 1;

        __syncthreads();   // all waves done reading buf[1-cur] (tile t-1)

        if (wave == 0) {
            if (t + 1 < ntiles) {
                const int kvn = kv_lo + (t + 1) * BN;
                const int nb  = (t + 1) & 1;
                tdm_load_2d_f16((unsigned int)(uintptr_t)&kLds[nb][0][0],
                                khw + (size_t)kvn * D_DIM,
                                D_DIM, S_LEN, D_DIM, D_DIM, BN);
                tdm_load_2d_f16((unsigned int)(uintptr_t)&vtLds[nb][0][0],
                                vhw + kvn,
                                S_LEN, D_DIM, S_LEN, BN, D_DIM);
                __builtin_amdgcn_s_wait_tensorcnt((short)2); // pair(t) complete
            } else {
                __builtin_amdgcn_s_wait_tensorcnt((short)0); // last pair complete
            }
        }
        __syncthreads();   // tile t data visible to all waves

        // ---- S = Q K^T : load 4 B-frags up-front, then 4 back-to-back WMMAs ----
        v8f sfrag[2];
        #pragma unroll
        for (int j = 0; j < 2; ++j) {
            V16 b[4];
            #pragma unroll
            for (int c = 0; c < 4; ++c) {
                const _Float16* src = &kLds[cur][j * 16 + l16][32 * c + (hi ? 8 : 0)];
                b[c].h[0] = *(const v8h*)src;
                b[c].h[1] = *(const v8h*)(src + 16);
            }
            v8f acc;
            #pragma unroll
            for (int r = 0; r < 8; ++r) acc[r] = 0.f;
            #pragma unroll
            for (int c = 0; c < 4; ++c)
                acc = __builtin_amdgcn_wmma_f32_16x16x32_f16(
                        false, qa[c].v, false, b[c].v, (short)0, acc, false, false);
            sfrag[j] = acc;
        }

        // ---- scale + block-causal mask ----
        #pragma unroll
        for (int j = 0; j < 2; ++j) {
            const int colp = kv + j * 16 + l16;
            int segc = 0;
            #pragma unroll
            for (int i = 1; i <= NDOC; ++i) if (colp >= off[i]) segc = i;
            #pragma unroll
            for (int r = 0; r < 8; ++r) {
                const float sv = sfrag[j][r] * scale;
                const bool ok  = (colp <= rbase + r) && (segc == segrow[r]);
                sfrag[j][r] = ok ? sv : NEG_INF;
            }
        }

        // ---- row max across the 16 lanes of this half-wave ----
        float rmax[8];
        #pragma unroll
        for (int r = 0; r < 8; ++r) rmax[r] = fmaxf(sfrag[0][r], sfrag[1][r]);
        #pragma unroll
        for (int dlt = 1; dlt < 16; dlt <<= 1)
            #pragma unroll
            for (int r = 0; r < 8; ++r)
                rmax[r] = fmaxf(rmax[r], __shfl_xor(rmax[r], dlt, 32));

        float alpha[8];
        #pragma unroll
        for (int r = 0; r < 8; ++r) {
            const float mnew = fmaxf(mrow[r], rmax[r]);
            alpha[r] = (mrow[r] > NEG_INF) ? __expf(mrow[r] - mnew) : 0.f;
            mrow[r]  = mnew;
            lrow[r] *= alpha[r];
        }

        // ---- p = exp(s - m); row sums ----
        float rsum[8];
        #pragma unroll
        for (int r = 0; r < 8; ++r) rsum[r] = 0.f;
        #pragma unroll
        for (int j = 0; j < 2; ++j)
            #pragma unroll
            for (int r = 0; r < 8; ++r) {
                const float sv = sfrag[j][r];
                const float p  = (sv > NEG_INF) ? __expf(sv - mrow[r]) : 0.f;
                sfrag[j][r] = p;
                rsum[r] += p;
            }
        #pragma unroll
        for (int dlt = 1; dlt < 16; dlt <<= 1)
            #pragma unroll
            for (int r = 0; r < 8; ++r)
                rsum[r] += __shfl_xor(rsum[r], dlt, 32);
        #pragma unroll
        for (int r = 0; r < 8; ++r) lrow[r] += rsum[r];

        // ---- rescale O accumulators ----
        #pragma unroll
        for (int dc = 0; dc < 8; ++dc)
            #pragma unroll
            for (int r = 0; r < 8; ++r) oacc[dc][r] *= alpha[r];

        // ---- P: C-layout -> per-wave LDS -> A-layout fragment ----
        #pragma unroll
        for (int j = 0; j < 2; ++j)
            #pragma unroll
            for (int r = 0; r < 8; ++r)
                pLds[wave][r + 8 * hi][j * 16 + l16] = (_Float16)sfrag[j][r];

        asm volatile("s_wait_dscnt 0" ::: "memory"); // same-wave store->load fence

        V16 pa;
        {
            const _Float16* ps = &pLds[wave][l16][hi ? 8 : 0];
            pa.h[0] = *(const v8h*)ps;
            pa.h[1] = *(const v8h*)(ps + 16);
        }

        // ---- O += P V : two groups of (4 frag loads, then 4 WMMAs) ----
        #pragma unroll
        for (int g = 0; g < 2; ++g) {
            V16 b[4];
            #pragma unroll
            for (int q4 = 0; q4 < 4; ++q4) {
                const int dc = g * 4 + q4;
                const _Float16* vs = &vtLds[cur][dc * 16 + l16][hi ? 8 : 0];
                b[q4].h[0] = *(const v8h*)vs;
                b[q4].h[1] = *(const v8h*)(vs + 16);
            }
            #pragma unroll
            for (int q4 = 0; q4 < 4; ++q4) {
                const int dc = g * 4 + q4;
                oacc[dc] = __builtin_amdgcn_wmma_f32_16x16x32_f16(
                             false, pa.v, false, b[q4].v, (short)0, oacc[dc],
                             false, false);
            }
        }
    }

    // ---- finalize: O / l ----
    #pragma unroll
    for (int r = 0; r < 8; ++r) {
        const float inv = (lrow[r] > 0.f) ? (1.f / lrow[r]) : 0.f;
        #pragma unroll
        for (int dc = 0; dc < 8; ++dc)
            oh[(size_t)(rbase + r) * D_DIM + dc * 16 + l16] = oacc[dc][r] * inv;
    }
}

// ---------------------------------------------------------------------------
// Fallback kernel (no workspace): in-kernel f32->f16 conversion, as round 1.
// ---------------------------------------------------------------------------
__global__ __launch_bounds__(NTHREADS)
void fa_fallback_kernel(const float* __restrict__ qg,
                        const float* __restrict__ kg,
                        const float* __restrict__ vg,
                        const long long* __restrict__ offsets,
                        float* __restrict__ outg)
{
    __shared__ _Float16 qLds[BM][D_DIM];
    __shared__ _Float16 kLds[BN][D_DIM];
    __shared__ _Float16 vtLds[D_DIM][BN];
    __shared__ _Float16 pLds[NTHREADS/32][16][BN];

    const int tid  = threadIdx.x;
    const int wave = tid >> 5;
    const int lane = tid & 31;
    const int l16  = lane & 15;
    const int hi   = lane >> 4;

    const int head = blockIdx.y;
    const int m0   = blockIdx.x * BM;

    const float* qh = qg + (size_t)head * S_LEN * D_DIM;
    const float* kh = kg + (size_t)head * S_LEN * D_DIM;
    const float* vh = vg + (size_t)head * S_LEN * D_DIM;
    float*       oh = outg + (size_t)head * S_LEN * D_DIM;

    int off[NDOC + 1];
    #pragma unroll
    for (int i = 0; i <= NDOC; ++i) off[i] = (int)offsets[i];

    for (int e = tid; e < BM * D_DIM; e += NTHREADS) {
        const int r = e >> 7, c = e & (D_DIM - 1);
        qLds[r][c] = (_Float16)qh[(size_t)(m0 + r) * D_DIM + c];
    }
    __syncthreads();

    const int wrow0 = wave * 16;
    V16 qa[4];
    {
        const int row = wrow0 + l16;
        const int dof = hi ? 8 : 0;
        #pragma unroll
        for (int c = 0; c < 4; ++c) {
            qa[c].h[0] = *(const v8h*)&qLds[row][32 * c + dof];
            qa[c].h[1] = *(const v8h*)&qLds[row][32 * c + 16 + dof];
        }
    }

    const int rbase = m0 + wrow0 + 8 * hi;
    int segrow[8];
    #pragma unroll
    for (int r = 0; r < 8; ++r) {
        int p = rbase + r, s = 0;
        #pragma unroll
        for (int i = 1; i <= NDOC; ++i) if (p >= off[i]) s = i;
        segrow[r] = s;
    }

    const float NEG_INF = -__builtin_inff();
    float mrow[8], lrow[8];
    v8f oacc[8];
    #pragma unroll
    for (int r = 0; r < 8; ++r) { mrow[r] = NEG_INF; lrow[r] = 0.f; }
    #pragma unroll
    for (int dc = 0; dc < 8; ++dc)
        #pragma unroll
        for (int r = 0; r < 8; ++r) oacc[dc][r] = 0.f;

    int seg0 = 0;
    #pragma unroll
    for (int i = 1; i <= NDOC; ++i) if (m0 >= off[i]) seg0 = i;
    const int kv_lo  = off[seg0] & ~(BN - 1);
    const int kv_end = m0 + BM;

    const float scale = 0.08838834764831845f;

    for (int kv = kv_lo; kv < kv_end; kv += BN) {
        __syncthreads();

        if (kv + BN < kv_end)
            __builtin_prefetch(&kh[(size_t)(kv + BN + lane) * D_DIM], 0, 0);

        for (int e = tid; e < BN * D_DIM; e += NTHREADS) {
            const int r = e >> 7, c = e & (D_DIM - 1);
            const size_t g = (size_t)(kv + r) * D_DIM + c;
            kLds[r][c]  = (_Float16)kh[g];
            vtLds[c][r] = (_Float16)vh[g];
        }
        __syncthreads();

        v8f sfrag[2];
        #pragma unroll
        for (int j = 0; j < 2; ++j) {
            V16 b[4];
            #pragma unroll
            for (int c = 0; c < 4; ++c) {
                const _Float16* src = &kLds[j * 16 + l16][32 * c + (hi ? 8 : 0)];
                b[c].h[0] = *(const v8h*)src;
                b[c].h[1] = *(const v8h*)(src + 16);
            }
            v8f acc;
            #pragma unroll
            for (int r = 0; r < 8; ++r) acc[r] = 0.f;
            #pragma unroll
            for (int c = 0; c < 4; ++c)
                acc = __builtin_amdgcn_wmma_f32_16x16x32_f16(
                        false, qa[c].v, false, b[c].v, (short)0, acc, false, false);
            sfrag[j] = acc;
        }

        #pragma unroll
        for (int j = 0; j < 2; ++j) {
            const int colp = kv + j * 16 + l16;
            int segc = 0;
            #pragma unroll
            for (int i = 1; i <= NDOC; ++i) if (colp >= off[i]) segc = i;
            #pragma unroll
            for (int r = 0; r < 8; ++r) {
                const float sv = sfrag[j][r] * scale;
                const bool ok  = (colp <= rbase + r) && (segc == segrow[r]);
                sfrag[j][r] = ok ? sv : NEG_INF;
            }
        }

        float rmax[8];
        #pragma unroll
        for (int r = 0; r < 8; ++r) rmax[r] = fmaxf(sfrag[0][r], sfrag[1][r]);
        #pragma unroll
        for (int dlt = 1; dlt < 16; dlt <<= 1)
            #pragma unroll
            for (int r = 0; r < 8; ++r)
                rmax[r] = fmaxf(rmax[r], __shfl_xor(rmax[r], dlt, 32));

        float alpha[8];
        #pragma unroll
        for (int r = 0; r < 8; ++r) {
            const float mnew = fmaxf(mrow[r], rmax[r]);
            alpha[r] = (mrow[r] > NEG_INF) ? __expf(mrow[r] - mnew) : 0.f;
            mrow[r]  = mnew;
            lrow[r] *= alpha[r];
        }

        float rsum[8];
        #pragma unroll
        for (int r = 0; r < 8; ++r) rsum[r] = 0.f;
        #pragma unroll
        for (int j = 0; j < 2; ++j)
            #pragma unroll
            for (int r = 0; r < 8; ++r) {
                const float sv = sfrag[j][r];
                const float p  = (sv > NEG_INF) ? __expf(sv - mrow[r]) : 0.f;
                sfrag[j][r] = p;
                rsum[r] += p;
            }
        #pragma unroll
        for (int dlt = 1; dlt < 16; dlt <<= 1)
            #pragma unroll
            for (int r = 0; r < 8; ++r)
                rsum[r] += __shfl_xor(rsum[r], dlt, 32);
        #pragma unroll
        for (int r = 0; r < 8; ++r) lrow[r] += rsum[r];

        #pragma unroll
        for (int dc = 0; dc < 8; ++dc)
            #pragma unroll
            for (int r = 0; r < 8; ++r) oacc[dc][r] *= alpha[r];

        #pragma unroll
        for (int j = 0; j < 2; ++j)
            #pragma unroll
            for (int r = 0; r < 8; ++r)
                pLds[wave][r + 8 * hi][j * 16 + l16] = (_Float16)sfrag[j][r];

        asm volatile("s_wait_dscnt 0" ::: "memory");

        V16 pa;
        {
            const _Float16* ps = &pLds[wave][l16][hi ? 8 : 0];
            pa.h[0] = *(const v8h*)ps;
            pa.h[1] = *(const v8h*)(ps + 16);
        }

        #pragma unroll
        for (int g = 0; g < 2; ++g) {
            V16 b[4];
            #pragma unroll
            for (int q4 = 0; q4 < 4; ++q4) {
                const int dc = g * 4 + q4;
                const _Float16* vs = &vtLds[dc * 16 + l16][hi ? 8 : 0];
                b[q4].h[0] = *(const v8h*)vs;
                b[q4].h[1] = *(const v8h*)(vs + 16);
            }
            #pragma unroll
            for (int q4 = 0; q4 < 4; ++q4) {
                const int dc = g * 4 + q4;
                oacc[dc] = __builtin_amdgcn_wmma_f32_16x16x32_f16(
                             false, pa.v, false, b[q4].v, (short)0, oacc[dc],
                             false, false);
            }
        }
    }

    #pragma unroll
    for (int r = 0; r < 8; ++r) {
        const float inv = (lrow[r] > 0.f) ? (1.f / lrow[r]) : 0.f;
        #pragma unroll
        for (int dc = 0; dc < 8; ++dc)
            oh[(size_t)(rbase + r) * D_DIM + dc * 16 + l16] = oacc[dc][r] * inv;
    }
}

extern "C" void kernel_launch(void* const* d_in, const int* in_sizes, int n_in,
                              void* d_out, int out_size, void* d_ws, size_t ws_size,
                              hipStream_t stream) {
    const float*     q       = (const float*)d_in[0];
    const float*     k       = (const float*)d_in[1];
    const float*     v       = (const float*)d_in[2];
    const long long* offsets = (const long long*)d_in[3];
    float*           out     = (float*)d_out;

    const size_t elems = (size_t)H_NUM * S_LEN * D_DIM;
    const size_t need  = 2 * elems * sizeof(_Float16);   // 16 MB: Kf16 + Vt f16
    dim3 grid(S_LEN / BM, H_NUM);

    if (d_ws != nullptr && ws_size >= need) {
        _Float16* kws  = (_Float16*)d_ws;
        _Float16* vtws = kws + elems;
        convert_kv_kernel<<<dim3((unsigned)(elems / 256)), 256, 0, stream>>>(k, v, kws, vtws);
        fa_tdm_kernel<<<grid, NTHREADS, 0, stream>>>(q, kws, vtws, offsets, out);
    } else {
        fa_fallback_kernel<<<grid, NTHREADS, 0, stream>>>(q, k, v, offsets, out);
    }
}
